// RNNEncoder_4801773437624
// MI455X (gfx1250) — matrix operand order repaired and checked by reading, not compiled, not run on hardware
//
#include <hip/hip_runtime.h>

// ---------- problem constants ----------
#define VOCAB 32000
#define EMBD  256
#define HID   512
#define BATCH 32
#define TSEQ  512
#define G4H   (4 * HID)      // 2048
#define NTILES (G4H / 16)    // 128 column tiles of 16
#define NBLK   32            // blocks in persistent recurrence

typedef __attribute__((ext_vector_type(16))) __bf16 v16bf;
typedef __attribute__((ext_vector_type(8)))  __bf16 v8bf;
typedef __attribute__((ext_vector_type(8)))  float  v8f;

// float -> bf16, round to nearest even
static __device__ __forceinline__ __bf16 f2bf(float f) {
    unsigned u = __builtin_bit_cast(unsigned, f);
    unsigned r = u + 0x7FFFu + ((u >> 16) & 1u);
    unsigned short h = (unsigned short)(r >> 16);
    return __builtin_bit_cast(__bf16, h);
}

static __device__ __forceinline__ float sigmoidf_fast(float x) {
    return 1.0f / (1.0f + __expf(-x));
}

// Build a 16x32 (MxK) bf16 A-fragment for v_wmma_f32_16x16x32_bf16 from a
// row-major matrix. Lane L holds row M = L%16; elements 0..15 hold
// K = (L/16)*16 .. +15 contiguously (matches packed B fragments).
static __device__ __forceinline__ v16bf load_a_frag(const __bf16* A,
                                                    int row, int lda, int kbase, int lane) {
    const __bf16* p = A + (size_t)row * lda + kbase + ((lane >> 4) << 4);
    v8bf a0 = *(const v8bf*)p;          // K = kh .. kh+7
    v8bf a1 = *(const v8bf*)(p + 8);    // K = kh+8 .. kh+15
    v16bf a;
#pragma unroll
    for (int j = 0; j < 8; ++j) { a[j] = a0[j]; a[j + 8] = a1[j]; }
    return a;
}

// ---------- kernel 1: pack weights into B-fragment layout ----------
// W: [K x 2048] row-major f32.  Fragment (ntile, kstep) = 32x16 bf16 tile;
// lane L stores column n = ntile*16 + (L&15), rows k = kstep*32 + (L>>4)*16
// + 0..15 contiguously (one v16bf = 32 bytes per lane).
__global__ void pack_w_kernel(const float* __restrict__ W,
                              __bf16* __restrict__ P, int ksteps) {
    int idx  = blockIdx.x * blockDim.x + threadIdx.x;   // (nt*ksteps + ks)*32 + lane
    int lane = idx & 31;
    int tile = idx >> 5;
    int ks   = tile % ksteps;
    int nt   = tile / ksteps;
    if (nt >= NTILES) return;
    int col   = nt * 16 + (lane & 15);
    int kbase = ks * 32 + ((lane >> 4) << 4);
    __bf16* dst = P + (size_t)idx * 16;
#pragma unroll
    for (int j = 0; j < 16; ++j)
        dst[j] = f2bf(W[(size_t)(kbase + j) * G4H + col]);
}

// ---------- kernel 2: embedding gather + bf16 convert ----------
__global__ void embed_kernel(const int* __restrict__ x,
                             const float* __restrict__ emb,
                             __bf16* __restrict__ xe) {
    int i   = blockIdx.x * blockDim.x + threadIdx.x;    // over B*T*EMBD
    int row = i >> 8;        // EMBD == 256
    int e   = i & 255;
    int tok = x[row];
    xe[i] = f2bf(emb[(size_t)tok * EMBD + e]);
}

// ---------- kernel 3: zero initial h (bf16), c (f32), barrier counter ----------
__global__ void zero_state_kernel(__bf16* __restrict__ h0, float* __restrict__ c,
                                  unsigned* __restrict__ barcnt) {
    int i = blockIdx.x * blockDim.x + threadIdx.x;      // B*HID = 16384
    h0[i] = f2bf(0.0f);
    c[i]  = 0.0f;
    if (i == 0) *barcnt = 0u;
}

// ---------- kernel 4: xg = xe @ Wx + b  ([16384x256]x[256x2048]) ----------
// 8 waves / block; each wave computes one 16x64 strip (4 N-tiles, reusing A).
__global__ void __launch_bounds__(256) xg_gemm_kernel(const __bf16* __restrict__ xe,
                                                      const __bf16* __restrict__ WxP,
                                                      const float* __restrict__ bias,
                                                      float* __restrict__ xg) {
    int lane   = threadIdx.x & 31;
    int wave   = threadIdx.x >> 5;
    int gw     = blockIdx.x * 8 + wave;
    int mtile  = gw >> 5;            // 1024 M-tiles
    int ngroup = gw & 31;            // 32 groups of 4 N-tiles
    int row    = mtile * 16 + (lane & 15);

    v8f acc[4];
#pragma unroll
    for (int i = 0; i < 4; ++i) acc[i] = (v8f){};

    const v16bf* bfrags = (const v16bf*)WxP;
    for (int ks = 0; ks < 8; ++ks) {                     // K = 256 = 8*32
        v16bf a = load_a_frag(xe, row, EMBD, ks * 32, lane);
#pragma unroll
        for (int i = 0; i < 4; ++i) {
            int nt = ngroup * 4 + i;
            v16bf b = bfrags[((size_t)nt * 8 + ks) * 32 + lane];
            acc[i] = __builtin_amdgcn_wmma_f32_16x16x32_bf16(
                false, a, false, b, (short)0, acc[i], false, false);
        }
    }

    // C/D layout: lane L -> col n = L&15, rows r + 8*(L>>4)
    int nloc  = lane & 15;
    int rbase = mtile * 16 + ((lane >> 4) << 3);
#pragma unroll
    for (int i = 0; i < 4; ++i) {
        int col  = (ngroup * 4 + i) * 16 + nloc;
        float bv = bias[col];
#pragma unroll
        for (int r = 0; r < 8; ++r)
            xg[(size_t)(rbase + r) * G4H + col] = acc[i][r] + bv;
    }
}

// ---------- kernel 5: persistent LSTM recurrence ----------
// grid = 32 blocks (one 16-col j-slice each), block = 256 threads (8 waves).
// Each block async-copies its 128 KB Wh fragment slice into LDS ONCE
// (global_load_async_to_lds_b128, ASYNCcnt), then iterates all 512 timesteps,
// synchronizing across blocks with an atomic grid barrier.  CDNA5's 320 KB
// LDS/workgroup makes the 136 KB residency possible (CDNA4 could not).
__global__ void __launch_bounds__(256) lstm_persistent_kernel(
        const float* __restrict__ xg,
        const __bf16* __restrict__ WhP,
        __bf16* __restrict__ h0buf,
        __bf16* __restrict__ h1buf,
        float* __restrict__ c,
        float* __restrict__ out,
        unsigned* __restrict__ barcnt) {
    // 8 tiles x 16 ksteps x (32 lanes x 16 bf16) = 128 KB of Wh fragments
    __shared__ __attribute__((aligned(32))) __bf16 whlds[8 * 16 * 32 * 16];
    __shared__ float zs[8 * 256];    // 8 tiles of 16x16 f32 gate staging

    int lane   = threadIdx.x & 31;
    int wave   = threadIdx.x >> 5;
    int mtile  = wave & 1;
    int gate   = wave >> 1;
    int jslice = blockIdx.x;
    int ntile  = gate * 32 + jslice;         // column tile in [0,128)
    int row    = mtile * 16 + (lane & 15);   // batch row for A

    // ---- stage Wh slice: memory -> LDS, no VGPR bounce (ASYNCcnt) ----
    for (int ks = 0; ks < 16; ++ks) {
        const __bf16* gsrc = WhP + ((size_t)(ntile * 16 + ks) * 32 + lane) * 16;
        // low 32 bits of a generic LDS address are the LDS byte offset
        unsigned loff = (unsigned)(size_t)(&whlds[((wave * 16 + ks) * 32 + lane) * 16]);
        asm volatile("global_load_async_to_lds_b128 %0, %1, off"
                     :: "v"(loff), "v"(gsrc) : "memory");
        asm volatile("global_load_async_to_lds_b128 %0, %1, off offset:16"
                     :: "v"(loff + 16u), "v"(gsrc) : "memory");
    }
    asm volatile("s_wait_asynccnt 0" ::: "memory");
    __syncthreads();

    const v16bf* bfrags = (const v16bf*)whlds;
    int nloc  = lane & 15;
    int col   = ntile * 16 + nloc;           // column in [0, 2048)
    int rhalf = (lane >> 4) << 3;

    for (int t = 0; t < TSEQ; ++t) {
        const __bf16* hp = (t & 1) ? h1buf : h0buf;
        __bf16*       hn = (t & 1) ? h0buf : h1buf;

        // ---- z-tile GEMM: A from global h (32 KB, L2-hot), B from LDS ----
        v8f acc = (v8f){};
        for (int ks = 0; ks < 16; ++ks) {              // K = 512 = 16*32
            v16bf a = load_a_frag(hp, row, HID, ks * 32, lane);
            v16bf b = bfrags[(wave * 16 + ks) * 32 + lane];
            acc = __builtin_amdgcn_wmma_f32_16x16x32_bf16(
                false, a, false, b, (short)0, acc, false, false);
        }

        // ---- add xg_t, stage z tile in LDS ----
#pragma unroll
        for (int r = 0; r < 8; ++r) {
            int brow = mtile * 16 + rhalf + r;         // batch in [0,32)
            float z  = acc[r] + xg[((size_t)brow * TSEQ + t) * G4H + col];
            zs[wave * 256 + (rhalf + r) * 16 + nloc] = z;
        }
        __syncthreads();

        // ---- fused gates: 512 elements (32 batch x 16 cols), 2/thread ----
        for (int e = threadIdx.x; e < 512; e += 256) {
            int brow = e >> 4;
            int ncol = e & 15;
            int mt   = brow >> 4;
            int rit  = brow & 15;
            float zi = zs[((0 * 2 + mt) * 256) + rit * 16 + ncol];
            float zf = zs[((1 * 2 + mt) * 256) + rit * 16 + ncol];
            float zg = zs[((2 * 2 + mt) * 256) + rit * 16 + ncol];
            float zo = zs[((3 * 2 + mt) * 256) + rit * 16 + ncol];

            float ig = sigmoidf_fast(zi);
            float fg = sigmoidf_fast(zf);
            float gg = tanhf(zg);
            float og = sigmoidf_fast(zo);

            int j = jslice * 16 + ncol;                // hidden index [0,512)
            size_t ci = (size_t)brow * HID + j;        // block-private
            float cn = fg * c[ci] + ig * gg;
            c[ci] = cn;
            float hn_f = og * tanhf(cn);
            out[((size_t)brow * TSEQ + t) * HID + j] = hn_f;
            hn[ci] = f2bf(hn_f);
        }

        // ---- grid barrier: publish h_t, wait for all 32 blocks ----
        __syncthreads();
        if (threadIdx.x == 0) {
            __threadfence();                           // release h writes
            atomicAdd(barcnt, 1u);
            unsigned target = (unsigned)NBLK * (unsigned)(t + 1);
            while (atomicAdd(barcnt, 0u) < target)
                __builtin_amdgcn_s_sleep(2);
        }
        __syncthreads();
        __threadfence();                               // acquire others' h
    }
}

// ---------- kernel 6: emit hT and cT tails ----------
__global__ void finalize_kernel(const float* __restrict__ out,
                                const float* __restrict__ c,
                                float* __restrict__ tail) {
    int i = blockIdx.x * blockDim.x + threadIdx.x;      // B*HID = 16384
    int b = i >> 9;
    int j = i & 511;
    tail[i] = out[((size_t)b * TSEQ + (TSEQ - 1)) * HID + j];   // hT
    tail[BATCH * HID + i] = c[i];                               // cT
}

// ---------- host launcher ----------
extern "C" void kernel_launch(void* const* d_in, const int* in_sizes, int n_in,
                              void* d_out, int out_size, void* d_ws, size_t ws_size,
                              hipStream_t stream) {
    const int*   x    = (const int*)d_in[0];
    const float* emb  = (const float*)d_in[1];
    const float* Wx   = (const float*)d_in[2];
    const float* Wh   = (const float*)d_in[3];
    const float* bias = (const float*)d_in[4];
    float* out = (float*)d_out;

    // workspace layout (all offsets 64 KB-aligned)
    char* ws = (char*)d_ws;
    __bf16* WxP = (__bf16*)(ws);                              // 1 MB  (128*8 frags)
    __bf16* WhP = (__bf16*)(ws + (1u << 20));                 // 2 MB  (128*16 frags)
    __bf16* xeb = (__bf16*)(ws + (3u << 20));                 // 8 MB  [16384 x 256] bf16
    float*  xg  = (float*) (ws + (11u << 20));                // 128 MB [16384 x 2048] f32
    __bf16* h0  = (__bf16*)(ws + (139u << 20));               // 32 KB
    __bf16* h1  = (__bf16*)(ws + (139u << 20) + 65536u);      // 32 KB
    float*  c   = (float*) (ws + (139u << 20) + 131072u);     // 64 KB
    unsigned* barcnt = (unsigned*)(ws + (140u << 20));        // 4 B

    // pack weights into WMMA B-fragment layout (bf16)
    pack_w_kernel<<<(NTILES * 8 * 32) / 256, 256, 0, stream>>>(Wx, WxP, 8);
    pack_w_kernel<<<(NTILES * 16 * 32) / 256, 256, 0, stream>>>(Wh, WhP, 16);

    // embedding gather + bf16 convert
    embed_kernel<<<(BATCH * TSEQ * EMBD) / 256, 256, 0, stream>>>(x, emb, xeb);

    // zero h0, c0, barrier counter (every call -> deterministic replay)
    zero_state_kernel<<<(BATCH * HID) / 256, 256, 0, stream>>>(h0, c, barcnt);

    // xg = xe @ Wx + b   (1024 M-tiles x 32 N-groups, 8 waves/block)
    xg_gemm_kernel<<<(1024 * 32) / 8, 256, 0, stream>>>(xeb, WxP, bias, xg);

    // persistent recurrence: one launch, grid-barrier between timesteps
    lstm_persistent_kernel<<<NBLK, 256, 0, stream>>>(xg, WhP, h0, h1, c, out, barcnt);

    // hT / cT tails of the tuple output
    finalize_kernel<<<(BATCH * HID) / 256, 256, 0, stream>>>(
        out, c, out + (size_t)BATCH * TSEQ * HID);
}
